// GQA_20444044329518
// MI455X (gfx1250) — matrix operand order repaired and checked by reading, not compiled, not run on hardware
//
#include <hip/hip_runtime.h>
#include <hip/hip_bf16.h>

typedef __bf16 bf16_t;
typedef __attribute__((ext_vector_type(16))) __bf16 v16bf;
typedef __attribute__((ext_vector_type(8)))  __bf16 v8bf;
typedef __attribute__((ext_vector_type(8)))  float   v8f;
typedef __attribute__((ext_vector_type(4)))  unsigned int v4u;
typedef int gqa_v4i __attribute__((vector_size(16)));   // matches builtin param type

static constexpr int  B_   = 4;
static constexpr int  T_   = 1024;
static constexpr int  D_   = 2048;
static constexpr int  NH   = 32;
static constexpr int  NKV  = 8;
static constexpr int  HD   = 64;
static constexpr long ROWS = (long)B_ * T_;        // 4096

#define AS1 __attribute__((address_space(1)))
#define AS3 __attribute__((address_space(3)))

// gfx1250 async global->LDS copy path (ASYNCcnt-tracked), guarded so the
// build cannot break on a toolchain/pass that lacks the builtin.
#if __has_builtin(__builtin_amdgcn_global_load_async_to_lds_b128)
#define GQA_ASYNC_COPY 1
#else
#define GQA_ASYNC_COPY 0
#endif

#if GQA_ASYNC_COPY
#if __has_builtin(__builtin_amdgcn_s_wait_asynccnt)
#define GQA_WAIT_ASYNC(n) __builtin_amdgcn_s_wait_asynccnt(n)
#else
#define GQA_WAIT_ASYNC(n) asm volatile("s_wait_asynccnt %0" ::"i"(n) : "memory")
#endif
#else
#define GQA_WAIT_ASYNC(n)
#endif

// 16-byte global -> LDS copy primitive.
__device__ __forceinline__ void gqa_cp16(const bf16_t* g, bf16_t* l) {
#if GQA_ASYNC_COPY
  __builtin_amdgcn_global_load_async_to_lds_b128((AS1 gqa_v4i*)(void*)g,
                                                 (AS3 gqa_v4i*)(void*)l,
                                                 /*offset=*/0, /*cpol=*/0);
#else
  *(v4u*)l = *(const v4u*)g;
#endif
}

// ---------------- WMMA helpers (CDNA5 16x16x32 bf16, f32 acc) ----------------

__device__ __forceinline__ v8f wmma_bf16(v16bf a, v16bf b, v8f c) {
  return __builtin_amdgcn_wmma_f32_16x16x32_bf16(false, a, false, b, (short)0, c,
                                                 false, false);
}

// A fragment 16x32 (M x K), row-major source, lda in elements.
// Layout (ISA 7.12.2): lanes 0-15 / 16-31 both hold M=lane&15;
// lane<16: elems[0..7]=K0..7, elems[8..15]=K16..23; lane>=16: +8 on K.
__device__ __forceinline__ v16bf load_a_frag(const bf16_t* __restrict__ A, int lda,
                                             int m0, int k0, int lane) {
  int row = m0 + (lane & 15);
  const bf16_t* p = A + (long)row * lda + k0 + ((lane >> 4) << 3);
  v8bf lo = *(const v8bf*)p;
  v8bf hi = *(const v8bf*)(p + 16);
  v16bf f;
#pragma unroll
  for (int i = 0; i < 8; ++i) { f[i] = lo[i]; f[i + 8] = hi[i]; }
  return f;
}

// B fragment 32x16 (K x N) from BT stored N-major (BT[n*ldb + k]).
// Layout: N=lane&15; lane<16: elems[0..7]=K0..7, elems[8..15]=K16..23; lane>=16: +8 on K.
__device__ __forceinline__ v16bf load_b_frag(const bf16_t* __restrict__ BT, int ldb,
                                             int n0, int k0, int lane) {
  const bf16_t* p = BT + (long)(n0 + (lane & 15)) * ldb + k0 + ((lane >> 4) << 3);
  v8bf lo = *(const v8bf*)p;
  v8bf hi = *(const v8bf*)(p + 16);
  v16bf f;
#pragma unroll
  for (int i = 0; i < 8; ++i) { f[i] = lo[i]; f[i + 8] = hi[i]; }
  return f;
}

__device__ __forceinline__ v8f zero8() {
  v8f z;
#pragma unroll
  for (int i = 0; i < 8; ++i) z[i] = 0.0f;
  return z;
}

// ---------------- elementwise prep kernels ----------------

__global__ void gqa_f32_to_bf16(const float* __restrict__ in,
                                bf16_t* __restrict__ out, long n) {
  long i = (long)blockIdx.x * blockDim.x + threadIdx.x;
  if (i < n) out[i] = (bf16_t)in[i];
}

// WT[n*K + k] = (bf16) W[k*N + n]
__global__ void gqa_transpose_w(const float* __restrict__ W,
                                bf16_t* __restrict__ WT, int K, int N) {
  long i = (long)blockIdx.x * blockDim.x + threadIdx.x;
  long total = (long)K * N;
  if (i >= total) return;
  int  k = (int)(i % K);
  long n = i / K;
  WT[n * (long)K + k] = (bf16_t)W[(long)k * N + n];
}

// vt[b][kv][d][t] = (bf16) vf[(b*T+t)*512 + kv*64 + d]
__global__ void gqa_v_transpose(const float* __restrict__ vf,
                                bf16_t* __restrict__ vt) {
  long i = (long)blockIdx.x * blockDim.x + threadIdx.x;
  long total = (long)B_ * NKV * HD * T_;
  if (i >= total) return;
  int t = (int)(i % T_); long r = i / T_;
  int d = (int)(r % HD); r /= HD;
  int kv = (int)(r % NKV);
  int b  = (int)(r / NKV);
  vt[i] = (bf16_t)vf[((long)(b * T_ + t) * NKV + kv) * HD + d];
}

// ---------------- WMMA GEMM: C[MxN] = A[MxK] * BT[NxK]^T ----------------
// block = 128 threads (4 waves); block tile 64x64, K-step 64.
// A+B panels are staged into LDS once per block via async global->LDS DMA
// (double buffered, ASYNCcnt-synchronized), removing the 4x cross-wave
// B-panel reload from L2. 8 WMMAs per wave per barrier pair.

__global__ __launch_bounds__(128) void gqa_gemm_bf16(const bf16_t* __restrict__ A,
                                                     const bf16_t* __restrict__ BT,
                                                     float* __restrict__ C,
                                                     int M, int N, int K) {
  __shared__ bf16_t sA[2][64 * 64];   // 8 KB per buffer
  __shared__ bf16_t sB[2][64 * 64];

  int tid  = threadIdx.x;
  int lane = tid & 31;
  int wave = tid >> 5;
  int m0b = blockIdx.y * 64;
  int n0b = blockIdx.x * 64;

  // copy mapping: 64 rows x 64 elems per tile = 512 16-byte chunks; 4/thread.
  auto stage = [&](int buf, int k0) {
#pragma unroll
    for (int j = 0; j < 4; ++j) {
      int c   = tid + 128 * j;
      int row = c >> 3;
      int kk  = (c & 7) * 8;
      gqa_cp16(A  + (long)(m0b + row) * K + k0 + kk, &sA[buf][row * 64 + kk]);
      gqa_cp16(BT + (long)(n0b + row) * K + k0 + kk, &sB[buf][row * 64 + kk]);
    }
  };

  v8f acc[4];
#pragma unroll
  for (int j = 0; j < 4; ++j) acc[j] = zero8();

  int niter = K / 64;
  stage(0, 0);
  for (int i = 0; i < niter; ++i) {
    int buf = i & 1;
    if (i + 1 < niter) {
      stage(buf ^ 1, (i + 1) * 64);
      GQA_WAIT_ASYNC(8);   // 8 just-issued ops may remain; current buffer done
    } else {
      GQA_WAIT_ASYNC(0);
    }
    __syncthreads();

    const bf16_t* Ab = &sA[buf][0];
    const bf16_t* Bb = &sB[buf][0];
#pragma unroll
    for (int kk = 0; kk < 64; kk += 32) {
      v16bf a = load_a_frag(Ab, 64, wave * 16, kk, lane);
#pragma unroll
      for (int j = 0; j < 4; ++j) {
        v16bf b = load_b_frag(Bb, 64, j * 16, kk, lane);
        acc[j] = wmma_bf16(a, b, acc[j]);
      }
    }
    __syncthreads();
  }

  int hf = lane >> 4, col = lane & 15;
  int m0 = m0b + wave * 16;
#pragma unroll
  for (int j = 0; j < 4; ++j)
#pragma unroll
    for (int r = 0; r < 8; ++r)
      C[(long)(m0 + r + 8 * hf) * N + n0b + j * 16 + col] = acc[j][r];
}

// ---------------- fused RMSNorm + RoPE, layout change to [b, h, t, d] bf16 -----
// one wave per (b, t, h); lane holds the RoPE pair (d, d+32)

__global__ __launch_bounds__(256) void gqa_norm_rope(const float* __restrict__ in,
                                                     const float* __restrict__ w,
                                                     const float* __restrict__ cosT,
                                                     const float* __restrict__ sinT,
                                                     bf16_t* __restrict__ out, int nh) {
  int lane = threadIdx.x & 31;
  long wid = (long)blockIdx.x * (blockDim.x >> 5) + (threadIdx.x >> 5);
  int h  = (int)(wid % nh);
  long bt = wid / nh;
  int t = (int)(bt % T_);
  int b = (int)(bt / T_);

  const float* p = in + (bt * nh + h) * HD;
  float x1 = p[lane];
  float x2 = p[lane + 32];
  float ss = x1 * x1 + x2 * x2;
#pragma unroll
  for (int m = 1; m < 32; m <<= 1) ss += __shfl_xor(ss, m, 32);
  float r = rsqrtf(ss * (1.0f / 64.0f) + 1e-6f);

  float c = cosT[t * 32 + lane];
  float s = sinT[t * 32 + lane];
  float a  = x1 * r * w[lane];
  float bb = x2 * r * w[lane + 32];
  float o1 = a * c - bb * s;
  float o2 = a * s + bb * c;

  bf16_t* q = out + (((long)(b * nh + h)) * T_ + t) * HD;
  q[lane]      = (bf16_t)o1;
  q[lane + 32] = (bf16_t)o2;
}

// ---------------- flash attention (causal, GQA group=4), all-WMMA --------------
// block = 256 (8 waves); each wave = 16 queries of one (b,h).
// grid = (B*NH, T/128). Per 32-key chunk: 4 WMMAs scores + 4 WMMAs P@V.

__global__ __launch_bounds__(256) void gqa_flash_attn(const bf16_t* __restrict__ Q,
                                                      const bf16_t* __restrict__ Kb,
                                                      const bf16_t* __restrict__ Vt,
                                                      bf16_t* __restrict__ Y) {
  __shared__ bf16_t lds_p[8][16 * 32];

  int lane = threadIdx.x & 31;
  int wave = threadIdx.x >> 5;
  int bh = blockIdx.x;
  int b = bh >> 5, h = bh & 31;
  int kv = h >> 2;
  int q0 = blockIdx.y * 128 + wave * 16;

  const bf16_t* qbase = Q  + ((long)bh * T_ + q0) * HD;
  const bf16_t* kbase = Kb + ((long)(b * NKV + kv) * T_) * HD;   // [t][d]
  const bf16_t* vbase = Vt + ((long)(b * NKV + kv) * HD) * T_;   // [d][t]

  v16bf aq0 = load_a_frag(qbase, HD, 0, 0, lane);
  v16bf aq1 = load_a_frag(qbase, HD, 0, 32, lane);

  v8f o[4];
#pragma unroll
  for (int j = 0; j < 4; ++j) o[j] = zero8();
  float mrow[8], lrow[8];
#pragma unroll
  for (int r = 0; r < 8; ++r) { mrow[r] = -3.0e38f; lrow[r] = 0.0f; }

  const float scale = 0.125f;   // 1/sqrt(64)
  int hf = lane >> 4, col = lane & 15;
  bf16_t* lp = &lds_p[wave][0];

  int kend = q0 + 16;
  for (int kc = 0; kc < kend; kc += 32) {
    // prefetch next key chunk of K (32x64 bf16) and V (64 rows) into caches
    if (kc + 32 < kend) {
      __builtin_prefetch(kbase + (long)(kc + 32) * HD + lane * HD, 0, 0);
      __builtin_prefetch(vbase + (long)(lane * 2) * T_ + kc + 32, 0, 0);
      __builtin_prefetch(vbase + (long)(lane * 2 + 1) * T_ + kc + 32, 0, 0);
    }

    // ---- scores S = Q @ K^T (two 16x16 tiles, HD=64 = 2 x K32) ----
    v8f s[2];
#pragma unroll
    for (int ts = 0; ts < 2; ++ts) {
      s[ts] = zero8();
      v16bf bk0 = load_b_frag(kbase, HD, kc + ts * 16, 0, lane);
      v16bf bk1 = load_b_frag(kbase, HD, kc + ts * 16, 32, lane);
      s[ts] = wmma_bf16(aq0, bk0, s[ts]);
      s[ts] = wmma_bf16(aq1, bk1, s[ts]);
    }

    // ---- scale + causal mask + per-row max (16-lane reduce within half) ----
    float pmax[8];
#pragma unroll
    for (int r = 0; r < 8; ++r) {
      int qrow = q0 + r + 8 * hf;
      float mx = -3.0e38f;
#pragma unroll
      for (int ts = 0; ts < 2; ++ts) {
        int kidx = kc + ts * 16 + col;
        float v = s[ts][r] * scale;
        v = (kidx <= qrow) ? v : -3.0e38f;
        s[ts][r] = v;
        mx = fmaxf(mx, v);
      }
      pmax[r] = mx;
    }
#pragma unroll
    for (int r = 0; r < 8; ++r)
#pragma unroll
      for (int m = 1; m < 16; m <<= 1)
        pmax[r] = fmaxf(pmax[r], __shfl_xor(pmax[r], m, 32));

    // ---- online softmax update ----
    float corr[8], lsum[8];
#pragma unroll
    for (int r = 0; r < 8; ++r) {
      float nm = fmaxf(mrow[r], pmax[r]);
      corr[r] = __expf(mrow[r] - nm);
      mrow[r] = nm;
      float sum = 0.0f;
#pragma unroll
      for (int ts = 0; ts < 2; ++ts) {
        float pv = __expf(s[ts][r] - nm);
        s[ts][r] = pv;
        sum += pv;
      }
      lsum[r] = sum;
    }
#pragma unroll
    for (int r = 0; r < 8; ++r)
#pragma unroll
      for (int m = 1; m < 16; m <<= 1) lsum[r] += __shfl_xor(lsum[r], m, 32);
#pragma unroll
    for (int r = 0; r < 8; ++r) lrow[r] = lrow[r] * corr[r] + lsum[r];
#pragma unroll
    for (int j = 0; j < 4; ++j)
#pragma unroll
      for (int r = 0; r < 8; ++r) o[j][r] *= corr[r];

    // ---- P: C-layout -> LDS -> A-fragment (same-wave DS ops are in-order) ----
#pragma unroll
    for (int r = 0; r < 8; ++r) {
      int row = r + 8 * hf;
      lp[row * 32 + col]      = (bf16_t)s[0][r];
      lp[row * 32 + 16 + col] = (bf16_t)s[1][r];
    }
    v16bf ap;
    {
      int m  = lane & 15;
      int kb = (lane >> 4) * 8;
      v8bf plo = *(const v8bf*)&lp[m * 32 + kb];
      v8bf phi = *(const v8bf*)&lp[m * 32 + kb + 16];
#pragma unroll
      for (int i = 0; i < 8; ++i) { ap[i] = plo[i]; ap[i + 8] = phi[i]; }
    }

    // ---- O += P @ V  (V^T fragments contiguous over key tokens) ----
#pragma unroll
    for (int j = 0; j < 4; ++j) {
      v16bf bv = load_b_frag(vbase, T_, j * 16, kc, lane);
      o[j] = wmma_bf16(ap, bv, o[j]);
    }
  }

  // ---- normalize and emit bf16 y[b*T + t, h*64 + d] ----
  long yrow = (long)b * T_ + q0;
#pragma unroll
  for (int j = 0; j < 4; ++j)
#pragma unroll
    for (int r = 0; r < 8; ++r) {
      int row = r + 8 * hf;
      float val = o[j][r] / lrow[r];
      Y[(yrow + row) * D_ + h * HD + j * 16 + col] = (bf16_t)val;
    }
}

// ---------------- host side ----------------

static inline long cdiv(long a, long b) { return (a + b - 1) / b; }

extern "C" void kernel_launch(void* const* d_in, const int* in_sizes, int n_in,
                              void* d_out, int out_size, void* d_ws, size_t ws_size,
                              hipStream_t stream) {
  const float* x    = (const float*)d_in[0];
  const float* Wq   = (const float*)d_in[1];
  const float* Wk   = (const float*)d_in[2];
  const float* Wv   = (const float*)d_in[3];
  const float* Wo   = (const float*)d_in[4];
  const float* qnw  = (const float*)d_in[5];
  const float* knw  = (const float*)d_in[6];
  const float* cosT = (const float*)d_in[7];
  const float* sinT = (const float*)d_in[8];

  char* ws = (char*)d_ws;
  size_t off = 0;
  auto carve = [&](size_t bytes) { void* p = ws + off; off += (bytes + 255) & ~(size_t)255; return p; };

  bf16_t* xb  = (bf16_t*)carve(ROWS * D_ * 2);                 // x bf16          16 MB
  bf16_t* WqT = (bf16_t*)carve((size_t)D_ * D_ * 2);           // Wq^T bf16        8 MB
  bf16_t* WkT = (bf16_t*)carve((size_t)(NKV*HD) * D_ * 2);     //                  2 MB
  bf16_t* WvT = (bf16_t*)carve((size_t)(NKV*HD) * D_ * 2);     //                  2 MB
  bf16_t* WoT = (bf16_t*)carve((size_t)D_ * D_ * 2);           //                  8 MB
  float*  qf  = (float*)carve(ROWS * D_ * 4);                  // q fp32          32 MB
  float*  kf  = (float*)carve(ROWS * (NKV*HD) * 4);            //                  8 MB
  float*  vf  = (float*)carve(ROWS * (NKV*HD) * 4);            //                  8 MB
  bf16_t* qb  = (bf16_t*)carve((size_t)B_*NH*T_*HD * 2);       // [b,h,t,d]       16 MB
  bf16_t* kb  = (bf16_t*)carve((size_t)B_*NKV*T_*HD * 2);      // [b,kv,t,d]       4 MB
  bf16_t* vt  = (bf16_t*)carve((size_t)B_*NKV*HD*T_ * 2);      // [b,kv,d,t]       4 MB
  bf16_t* yb  = (bf16_t*)carve(ROWS * D_ * 2);                 // attn out bf16   16 MB

  // 1) precision conversion / weight transposes
  long nx = ROWS * D_;
  gqa_f32_to_bf16<<<cdiv(nx, 256), 256, 0, stream>>>(x, xb, nx);
  gqa_transpose_w<<<cdiv((long)D_ * D_, 256), 256, 0, stream>>>(Wq, WqT, D_, D_);
  gqa_transpose_w<<<cdiv((long)D_ * NKV*HD, 256), 256, 0, stream>>>(Wk, WkT, D_, NKV*HD);
  gqa_transpose_w<<<cdiv((long)D_ * NKV*HD, 256), 256, 0, stream>>>(Wv, WvT, D_, NKV*HD);
  gqa_transpose_w<<<cdiv((long)D_ * D_, 256), 256, 0, stream>>>(Wo, WoT, D_, D_);

  // 2) projections (WMMA GEMMs, async-LDS staged)
  gqa_gemm_bf16<<<dim3(D_ / 64, ROWS / 64), 128, 0, stream>>>(xb, WqT, qf, ROWS, D_, D_);
  gqa_gemm_bf16<<<dim3((NKV*HD) / 64, ROWS / 64), 128, 0, stream>>>(xb, WkT, kf, ROWS, NKV*HD, D_);
  gqa_gemm_bf16<<<dim3((NKV*HD) / 64, ROWS / 64), 128, 0, stream>>>(xb, WvT, vf, ROWS, NKV*HD, D_);

  // 3) RMSNorm + RoPE + layout change (one wave per token-head)
  long qwaves = (long)B_ * T_ * NH;
  gqa_norm_rope<<<qwaves / 8, 256, 0, stream>>>(qf, qnw, cosT, sinT, qb, NH);
  long kwaves = (long)B_ * T_ * NKV;
  gqa_norm_rope<<<kwaves / 8, 256, 0, stream>>>(kf, knw, cosT, sinT, kb, NKV);

  // 4) V transpose to [b,kv,d,t]
  long nv = (long)B_ * NKV * HD * T_;
  gqa_v_transpose<<<cdiv(nv, 256), 256, 0, stream>>>(vf, vt);

  // 5) causal flash attention (WMMA)
  gqa_flash_attn<<<dim3(B_ * NH, T_ / 128), 256, 0, stream>>>(qb, kb, vt, yb);

  // 6) output projection (WMMA GEMM) -> fp32 d_out
  gqa_gemm_bf16<<<dim3(D_ / 64, ROWS / 64), 128, 0, stream>>>(yb, WoT, (float*)d_out,
                                                              ROWS, D_, D_);
}
